// Network_Triplane_68745246539840
// MI455X (gfx1250) — matrix operand order repaired and checked by reading, not compiled, not run on hardware
//
#include <hip/hip_runtime.h>
#include <hip/hip_bf16.h>
#include <hip/hip_fp16.h>

typedef __attribute__((ext_vector_type(2)))  float    v2f;
typedef __attribute__((ext_vector_type(8)))  float    v8f;
typedef __attribute__((ext_vector_type(16))) _Float16 v16h;
typedef __attribute__((ext_vector_type(8)))  _Float16 v8h;

#define N_RAYS 1024
#define N_SAMP 64
#define NPTS   (N_RAYS * N_SAMP)   // 65536
#define NV     6890
#define NVT    432                 // vertex tiles processed (EVEN -> clean 2x unroll)
#define NVTAB  6944                // (NVT+1)*16 rounded up; pad rows = +huge
#define FDIM   96
#define WDIM   128
#define KS     136                 // padded LDS row stride (f16 elems) -> conflict-free
#define MT     64                  // points per MLP block
#define MPW    32                  // points per NN wave (2 A-fragments)

// ---------------------------------------------------------------------------
// Prep: vertex table [-2x,-2y,-2z,|v|^2] (pad rows = +huge, never argmin) and
// transpose+convert W0/W1/W2 to f16 [N][K] so B-fragments are contiguous.
// ---------------------------------------------------------------------------
__global__ __launch_bounds__(256) void prep_kernel(
    const float* __restrict__ verts,
    const float* __restrict__ W0, const float* __restrict__ W1,
    const float* __restrict__ W2,
    float4* __restrict__ vtab,
    _Float16* __restrict__ w0t, _Float16* __restrict__ w1t,
    _Float16* __restrict__ w2t) {
  int i = blockIdx.x * blockDim.x + threadIdx.x;
  if (i < NVTAB) {
    float4 r;
    if (i < NV) {
      float x = verts[i * 3 + 0], y = verts[i * 3 + 1], z = verts[i * 3 + 2];
      r = make_float4(-2.f * x, -2.f * y, -2.f * z, x * x + y * y + z * z);
    } else {
      r = make_float4(0.f, 0.f, 0.f, 3.0e38f);
    }
    vtab[i] = r;
  }
  int j = i - NVTAB;
  if (j >= 0 && j < WDIM * FDIM) {            // W0t[n][k] = W0[k][n]
    int n = j / FDIM, k = j % FDIM;
    w0t[j] = (_Float16)W0[k * WDIM + n];
  }
  j -= WDIM * FDIM;
  if (j >= 0 && j < WDIM * WDIM) {            // W1t
    int n = j / WDIM, k = j % WDIM;
    w1t[j] = (_Float16)W1[k * WDIM + n];
  }
  j -= WDIM * WDIM;
  if (j >= 0 && j < WDIM * WDIM) {            // W2t
    int n = j / WDIM, k = j % WDIM;
    w2t[j] = (_Float16)W2[k * WDIM + n];
  }
}

// ---------------------------------------------------------------------------
// Per-ray min/max normalization; emit homogeneous point [x,y,z,1] (float4).
// ---------------------------------------------------------------------------
__global__ __launch_bounds__(64) void norm_kernel(
    const float* __restrict__ sp, float4* __restrict__ pts) {
  int ray = blockIdx.x, s = threadIdx.x;
  const float* p = sp + (size_t)(ray * N_SAMP + s) * 3;
  float x = p[0], y = p[1], z = p[2];
  float mnx = x, mny = y, mnz = z, mxx = x, mxy = y, mxz = z;
  #pragma unroll
  for (int m = 1; m < 32; m <<= 1) {
    mnx = fminf(mnx, __shfl_xor(mnx, m, 32));
    mny = fminf(mny, __shfl_xor(mny, m, 32));
    mnz = fminf(mnz, __shfl_xor(mnz, m, 32));
    mxx = fmaxf(mxx, __shfl_xor(mxx, m, 32));
    mxy = fmaxf(mxy, __shfl_xor(mxy, m, 32));
    mxz = fmaxf(mxz, __shfl_xor(mxz, m, 32));
  }
  __shared__ float red[2][6];
  int w = s >> 5, l = s & 31;
  if (l == 0) {
    red[w][0] = mnx; red[w][1] = mny; red[w][2] = mnz;
    red[w][3] = mxx; red[w][4] = mxy; red[w][5] = mxz;
  }
  __syncthreads();
  float Mnx = fminf(red[0][0], red[1][0]);
  float Mny = fminf(red[0][1], red[1][1]);
  float Mnz = fminf(red[0][2], red[1][2]);
  float Mxx = fmaxf(red[0][3], red[1][3]);
  float Mxy = fmaxf(red[0][4], red[1][4]);
  float Mxz = fmaxf(red[0][5], red[1][5]);
  pts[ray * N_SAMP + s] = make_float4((x - Mnx) / (Mxx - Mnx),
                                      (y - Mny) / (Mxy - Mny),
                                      (z - Mnz) / (Mxz - Mnz), 1.0f);
}

// ---------------------------------------------------------------------------
// Nearest-vertex argmin via WMMA f32 16x16x4 (homogeneous-coordinate trick):
//   D[M][N] = d^2 - |p|^2  (per-point constant offset -> argmin unchanged)
// 32 points/wave (two A fragments -> 2 WMMAs per vertex tile); lanes load only
// their K-half (8B); next tile register-prefetched; even trip count, unroll 2.
// ---------------------------------------------------------------------------
__global__ __launch_bounds__(256) void nn_kernel(
    const float4* __restrict__ pts, const float4* __restrict__ vtab,
    int* __restrict__ idx) {
  int tid = threadIdx.x;
  int wave = tid >> 5, lane = tid & 31;
  int ln = lane & 15;
  bool hi = lane >= 16;
  int hioff = hi ? 8 : 0;  // byte offset to this lane's K-half within a float4
  int mbase = (blockIdx.x * 8 + wave) * MPW;

  // A fragments: lane<16 -> (x,y) K=0,1 ; lane>=16 -> (z,1) K=2,3
  const char* pbase = (const char*)(pts + mbase) + hioff;
  v2f a0 = *(const v2f*)(pbase + ln * 16);
  v2f a1 = *(const v2f*)(pbase + (16 + ln) * 16);

  float best[16];
  int bidx[16];
  #pragma unroll
  for (int v = 0; v < 16; ++v) { best[v] = 3.4e38f; bidx[v] = 0; }

  const char* vbase = (const char*)vtab + ln * 16 + hioff;
  v2f b = *(const v2f*)(vbase);  // tile 0
  #pragma unroll 2
  for (int nt = 0; nt < NVT; ++nt) {
    v2f bn = *(const v2f*)(vbase + (size_t)(nt + 1) * 256);  // prefetch (padded)
    v8f c0, c1;
    #pragma unroll
    for (int v = 0; v < 8; ++v) { c0[v] = 0.f; c1[v] = 0.f; }
    c0 = __builtin_amdgcn_wmma_f32_16x16x4_f32(false, a0, false, b,
                                               (short)0, c0, false, false);
    c1 = __builtin_amdgcn_wmma_f32_16x16x4_f32(false, a1, false, b,
                                               (short)0, c1, false, false);
    int n = nt * 16 + ln;  // this lane's column = vertex index
    #pragma unroll
    for (int v = 0; v < 8; ++v) {
      float d0 = c0[v], d1 = c1[v];
      if (d0 < best[v])     { best[v] = d0;     bidx[v] = n; }
      if (d1 < best[8 + v]) { best[8 + v] = d1; bidx[8 + v] = n; }
    }
    b = bn;
  }
  // branchless butterfly min+index across the 16 lanes of each half
  #pragma unroll
  for (int v = 0; v < 16; ++v) {
    float bv = best[v];
    int bi = bidx[v];
    #pragma unroll
    for (int m = 1; m < 16; m <<= 1) {
      float ov = __shfl_xor(bv, m, 32);
      int oi = __shfl_xor(bi, m, 32);
      int take = (int)(ov < bv) | ((int)(ov == bv) & (int)(oi < bi));
      bv = take ? ov : bv;
      bi = take ? oi : bi;
    }
    bidx[v] = bi;
  }
  if (ln == 0) {  // lane 0 -> M=0..7 halves, lane 16 -> M=8..15 halves
    int mo = mbase + (hi ? 8 : 0);
    #pragma unroll
    for (int v = 0; v < 8; ++v) idx[mo + v] = bidx[v];
    #pragma unroll
    for (int v = 0; v < 8; ++v) idx[mo + 16 + v] = bidx[8 + v];
  }
}

// ---------------------------------------------------------------------------
// Gather + 3-layer MLP (f16 WMMA, f32 accum, fused bias+ReLU) + head + blend.
// 64 points/block, 8 waves; wave w owns output columns [16w,16w+16).
// Layer loop is k-outer / Mtile-inner: each weight fragment feeds 4 WMMAs.
// ---------------------------------------------------------------------------
__global__ __launch_bounds__(256) void mlp_kernel(
    const int* __restrict__ idx, const float* __restrict__ tf,
    const _Float16* __restrict__ w0t, const float* __restrict__ b0,
    const _Float16* __restrict__ w1t, const float* __restrict__ b1,
    const _Float16* __restrict__ w2t, const float* __restrict__ b2,
    const float* __restrict__ W3, const float* __restrict__ b3,
    const float* __restrict__ nerf, const float* __restrict__ bwv,
    float* __restrict__ out) {
  __shared__ __align__(32) _Float16 bufA[MT * KS];
  __shared__ __align__(32) _Float16 bufB[MT * KS];
  int tid = threadIdx.x;
  int base = blockIdx.x * MT;

  // gather: thread -> (point p = tid/4, quarter-row q = tid%4, 24 feats)
  {
    int p = tid >> 2, q = tid & 3;
    int row = idx[base + p];
    const float4* src = (const float4*)(tf + (size_t)row * FDIM + q * 24);
    _Float16 tmp[24];
    #pragma unroll
    for (int c = 0; c < 6; ++c) {
      float4 f = src[c];
      tmp[c * 4 + 0] = (_Float16)f.x;
      tmp[c * 4 + 1] = (_Float16)f.y;
      tmp[c * 4 + 2] = (_Float16)f.z;
      tmp[c * 4 + 3] = (_Float16)f.w;
    }
    v8h* dst = (v8h*)(bufA + p * KS + q * 24);  // 272B pitch + 48B -> 16B aligned
    #pragma unroll
    for (int c = 0; c < 3; ++c) {
      v8h vv;
      #pragma unroll
      for (int e = 0; e < 8; ++e) vv[e] = tmp[c * 8 + e];
      dst[c] = vv;
    }
  }
  __syncthreads();

  int wave = tid >> 5, lane = tid & 31, ln = lane & 15;
  bool hih = lane >= 16;
  int col = wave * 16 + ln;  // this lane's output column (B-frag N)

  auto layer = [&](const _Float16* A, int Kdim, const _Float16* Wt,
                   const float* bias, _Float16* O) {
    float bcol = bias[col];
    const _Float16* wcol = Wt + (size_t)col * Kdim + (hih ? 16 : 0);
    const _Float16* arow = A + ln * KS + (hih ? 16 : 0);
    v8f acc[4];
    #pragma unroll
    for (int mt = 0; mt < 4; ++mt)
      #pragma unroll
      for (int v = 0; v < 8; ++v) acc[mt][v] = bcol;  // bias broadcast over M
    #pragma unroll
    for (int k0 = 0; k0 < Kdim; k0 += 32) {
      v16h bf = *(const v16h*)(wcol + k0);  // one weight frag -> 4 WMMAs
      #pragma unroll
      for (int mt = 0; mt < 4; ++mt) {
        const _Float16* ap = arow + mt * 16 * KS + k0;
        union { v16h v; v8h h[2]; } au;
        au.h[0] = *(const v8h*)ap;
        au.h[1] = *(const v8h*)(ap + 8);
        acc[mt] = __builtin_amdgcn_wmma_f32_16x16x32_f16(
            false, au.v, false, bf, (short)0, acc[mt], false, false);
      }
    }
    #pragma unroll
    for (int mt = 0; mt < 4; ++mt)
      #pragma unroll
      for (int v = 0; v < 8; ++v) {  // ReLU + scatter to next activation tile
        float r = acc[mt][v] > 0.f ? acc[mt][v] : 0.f;
        int mo = mt * 16 + v + (hih ? 8 : 0);
        O[mo * KS + col] = (_Float16)r;
      }
  };

  layer(bufA, FDIM, w0t, b0, bufB); __syncthreads();
  layer(bufB, WDIM, w1t, b1, bufA); __syncthreads();
  layer(bufA, WDIM, w2t, b2, bufB); __syncthreads();  // head reads all waves' cols

  // head: density[m] = h3[m,:] . W3 + b3 ; sigmoid-blend with nerf density
  int m = tid >> 2, q = tid & 3;
  const v8h* hrow = (const v8h*)(bufB + m * KS + q * 32);
  const float4* w4 = (const float4*)(W3 + q * 32);
  float s = 0.f;
  #pragma unroll
  for (int c = 0; c < 4; ++c) {
    v8h hv = hrow[c];
    float4 wa = w4[2 * c], wb = w4[2 * c + 1];
    s += (float)hv[0] * wa.x + (float)hv[1] * wa.y +
         (float)hv[2] * wa.z + (float)hv[3] * wa.w;
    s += (float)hv[4] * wb.x + (float)hv[5] * wb.y +
         (float)hv[6] * wb.z + (float)hv[7] * wb.w;
  }
  s += __shfl_xor(s, 1, 32);
  s += __shfl_xor(s, 2, 32);
  if (q == 0) {
    float bw = 1.f / (1.f + __expf(-bwv[0]));
    int g = base + m;
    out[g] = bw * nerf[g] + (1.f - bw) * (s + b3[0]);
  }
}

// ---------------------------------------------------------------------------
extern "C" void kernel_launch(void* const* d_in, const int* in_sizes, int n_in,
                              void* d_out, int out_size, void* d_ws, size_t ws_size,
                              hipStream_t stream) {
  const float* sp    = (const float*)d_in[0];   // [1024,64,3]
  const float* verts = (const float*)d_in[1];   // [6890,3]
  const float* tf    = (const float*)d_in[2];   // [6890,96]
  const float* nerf  = (const float*)d_in[3];   // [1024,64]
  const float* bwv   = (const float*)d_in[4];   // [1]
  const float* W0 = (const float*)d_in[5];  const float* b0 = (const float*)d_in[6];
  const float* W1 = (const float*)d_in[7];  const float* b1 = (const float*)d_in[8];
  const float* W2 = (const float*)d_in[9];  const float* b2 = (const float*)d_in[10];
  const float* W3 = (const float*)d_in[11]; const float* b3 = (const float*)d_in[12];
  // d_in[13] = chunk_size (irrelevant on GPU)

  char* ws = (char*)d_ws;
  const size_t off_pts  = 0;                              // 65536 * 16 B
  const size_t off_vtab = off_pts  + (size_t)NPTS * 16;   // NVTAB * 16 B
  const size_t off_idx  = off_vtab + (size_t)NVTAB * 16;  // 65536 * 4 B
  const size_t off_w0t  = off_idx  + (size_t)NPTS * 4;    // f16 128x96
  const size_t off_w1t  = off_w0t  + (size_t)WDIM * FDIM * 2;
  const size_t off_w2t  = off_w1t  + (size_t)WDIM * WDIM * 2;

  float4*   pts  = (float4*)(ws + off_pts);
  float4*   vtab = (float4*)(ws + off_vtab);
  int*      idx  = (int*)(ws + off_idx);
  _Float16* w0t  = (_Float16*)(ws + off_w0t);
  _Float16* w1t  = (_Float16*)(ws + off_w1t);
  _Float16* w2t  = (_Float16*)(ws + off_w2t);

  const int prep_total = NVTAB + WDIM * FDIM + 2 * WDIM * WDIM;
  prep_kernel<<<(prep_total + 255) / 256, 256, 0, stream>>>(
      verts, W0, W1, W2, vtab, w0t, w1t, w2t);
  norm_kernel<<<N_RAYS, N_SAMP, 0, stream>>>(sp, pts);
  nn_kernel<<<NPTS / (8 * MPW), 256, 0, stream>>>(pts, vtab, idx);
  mlp_kernel<<<NPTS / MT, 256, 0, stream>>>(idx, tf, w0t, b0, w1t, b1, w2t, b2,
                                            W3, b3, nerf, bwv, (float*)d_out);
}
